// VisNetEncoder_26542897889595
// MI455X (gfx1250) — matrix operand order repaired and checked by reading, not compiled, not run on hardware
//
#include <hip/hip_runtime.h>
#include <math.h>

// ---------------- constants matching the reference ----------------
#define Hh    256
#define NHEAD 8
#define HDIM  32
#define LNUM  6
#define RNUM  32

typedef __attribute__((ext_vector_type(16))) __bf16 bf16x16;
typedef __attribute__((ext_vector_type(8)))  float  f32x8;

__device__ __forceinline__ float silu_f(float x) {
    return x / (1.0f + __expf(-x));
}

// =====================================================================
// Generic WMMA GEMM:  C[M,N] = act( A[M,K] @ B[N,K]^T + bias )
// A, B row-major f32; converted to bf16 on the fly.
// Each wave computes a 32x32 tile: 2 A fragments x 2 B fragments ->
// 4 v_wmma per K-step, each fragment reused twice (~8 FLOP/B from cache).
// Block = 4 waves -> 32 rows x 128 cols per block. N must be a multiple
// of 128 (true for all uses: 256, 512, 768). K templated: 32/256/512.
// M tail handled by clamping A rows (stores are guarded).
// act: 0 = none, 1 = SiLU
// =====================================================================
template <int K>
__device__ __forceinline__ bf16x16 load_a_frag(const float* __restrict__ Arow,
                                               int k0, int h) {
    const float4 A0 = *(const float4*)(Arow + k0 + 8 * h);
    const float4 A1 = *(const float4*)(Arow + k0 + 8 * h + 4);
    const float4 A2 = *(const float4*)(Arow + k0 + 16 + 8 * h);
    const float4 A3 = *(const float4*)(Arow + k0 + 16 + 8 * h + 4);
    bf16x16 a;
    a[0]  = (__bf16)A0.x;  a[1]  = (__bf16)A0.y;
    a[2]  = (__bf16)A0.z;  a[3]  = (__bf16)A0.w;
    a[4]  = (__bf16)A1.x;  a[5]  = (__bf16)A1.y;
    a[6]  = (__bf16)A1.z;  a[7]  = (__bf16)A1.w;
    a[8]  = (__bf16)A2.x;  a[9]  = (__bf16)A2.y;
    a[10] = (__bf16)A2.z;  a[11] = (__bf16)A2.w;
    a[12] = (__bf16)A3.x;  a[13] = (__bf16)A3.y;
    a[14] = (__bf16)A3.z;  a[15] = (__bf16)A3.w;
    return a;
}

template <int K>
__global__ void __launch_bounds__(128)
gemm_bf16_kernel(const float* __restrict__ A, const float* __restrict__ B,
                 const float* __restrict__ bias, float* __restrict__ C,
                 int M, int N, int act) {
    const int wave   = threadIdx.x >> 5;
    const int lane   = threadIdx.x & 31;
    const int m_base = blockIdx.x * 32;                // 32 rows per block
    const int n_base = (blockIdx.y * 4 + wave) * 32;   // 32 cols per wave

    const int h  = lane >> 4;       // lane half
    const int ml = lane & 15;
    int m0 = m_base + ml;           // A rows this lane supplies
    int m1 = m_base + 16 + ml;
    if (m0 >= M) m0 = M - 1;        // clamp: garbage rows never stored
    if (m1 >= M) m1 = M - 1;
    const float* Arow0 = A + (size_t)m0 * K;
    const float* Arow1 = A + (size_t)m1 * K;
    const float* B0    = B + (size_t)n_base * K;

    f32x8 acc00 = {}, acc01 = {}, acc10 = {}, acc11 = {};
    for (int k0 = 0; k0 < K; k0 += 32) {
        // ---- A fragments: two contiguous 8-float runs per lane each ----
        bf16x16 a0 = load_a_frag<K>(Arow0, k0, h);
        bf16x16 a1 = load_a_frag<K>(Arow1, k0, h);

        // ---- B fragments: lane = n-col; elements over k in table order ----
        const int kb = k0 + 8 * h;
        bf16x16 b0, b1;
        #pragma unroll
        for (int r = 0; r < 8; ++r) {
            const int kr = kb + 2 * r + (r >= 4 ? 8 : 0);
            b0[2 * r]     = (__bf16)B0[(size_t)ml * K + kr];
            b0[2 * r + 1] = (__bf16)B0[(size_t)ml * K + kr + 1];
            b1[2 * r]     = (__bf16)B0[(size_t)(16 + ml) * K + kr];
            b1[2 * r + 1] = (__bf16)B0[(size_t)(16 + ml) * K + kr + 1];
        }
        acc00 = __builtin_amdgcn_wmma_f32_16x16x32_bf16(
            false, a0, false, b0, (short)0, acc00, false, false);
        acc01 = __builtin_amdgcn_wmma_f32_16x16x32_bf16(
            false, a0, false, b1, (short)0, acc01, false, false);
        acc10 = __builtin_amdgcn_wmma_f32_16x16x32_bf16(
            false, a1, false, b0, (short)0, acc10, false, false);
        acc11 = __builtin_amdgcn_wmma_f32_16x16x32_bf16(
            false, a1, false, b1, (short)0, acc11, false, false);
    }
    // C/D layout: VGPR v -> M = v + 8*half ; N = lane & 15
    const int nn0 = n_base + ml;
    const int nn1 = n_base + 16 + ml;
    float bi0 = 0.0f, bi1 = 0.0f;
    if (bias) { bi0 = bias[nn0]; bi1 = bias[nn1]; }
    #pragma unroll
    for (int v = 0; v < 8; ++v) {
        int mmA = m_base + v + 8 * h;
        int mmB = m_base + 16 + v + 8 * h;
        if (mmA < M) {
            float v0 = acc00[v] + bi0;
            float v1 = acc01[v] + bi1;
            if (act) { v0 = silu_f(v0); v1 = silu_f(v1); }
            C[(size_t)mmA * N + nn0] = v0;
            C[(size_t)mmA * N + nn1] = v1;
        }
        if (mmB < M) {
            float v0 = acc10[v] + bi0;
            float v1 = acc11[v] + bi1;
            if (act) { v0 = silu_f(v0); v1 = silu_f(v1); }
            C[(size_t)mmB * N + nn0] = v0;
            C[(size_t)mmB * N + nn1] = v1;
        }
    }
}

static void launch_gemm(const float* A, const float* B, const float* bias,
                        float* C, int M, int N, int K, int act,
                        hipStream_t stream) {
    dim3 g((M + 31) / 32, N / 128);
    if (K == 32)
        gemm_bf16_kernel<32><<<g, 128, 0, stream>>>(A, B, bias, C, M, N, act);
    else if (K == 256)
        gemm_bf16_kernel<256><<<g, 128, 0, stream>>>(A, B, bias, C, M, N, act);
    else
        gemm_bf16_kernel<512><<<g, 128, 0, stream>>>(A, B, bias, C, M, N, act);
}

// =====================================================================
// Edge geometry: unit vector -> spherical basis, RBF, cosine cutoff
// =====================================================================
__global__ void edge_geom_kernel(const float* __restrict__ pos,
                                 const int* __restrict__ eidx, int E,
                                 float* __restrict__ sph,
                                 float* __restrict__ rbf,
                                 float* __restrict__ rcut) {
    int e = blockIdx.x * blockDim.x + threadIdx.x;
    if (e >= E) return;
    int s = eidx[e], d = eidx[E + e];
    float rx = pos[s * 3 + 0] - pos[d * 3 + 0];
    float ry = pos[s * 3 + 1] - pos[d * 3 + 1];
    float rz = pos[s * 3 + 2] - pos[d * 3 + 2];
    bool loop = (s == d);
    float d2 = rx * rx + ry * ry + rz * rz;
    float ew = loop ? 0.0f : sqrtf(d2);
    float inv = loop ? 0.0f : (1.0f / ew);
    float ux = rx * inv, uy = ry * inv, uz = rz * inv;
    const float s3 = 1.7320508075688772f;
    sph[e * 8 + 0] = ux;
    sph[e * 8 + 1] = uy;
    sph[e * 8 + 2] = uz;
    sph[e * 8 + 3] = s3 * ux * uz;
    sph[e * 8 + 4] = s3 * ux * uy;
    sph[e * 8 + 5] = uy * uy - 0.5f * (ux * ux + uz * uz);
    sph[e * 8 + 6] = s3 * uy * uz;
    sph[e * 8 + 7] = 0.5f * s3 * (uz * uz - ux * ux);
    float rc = (ew < 5.0f) ? 0.5f * (__cosf(ew * 0.6283185307179586f) + 1.0f) : 0.0f;
    rcut[e] = rc;
    const float START_ = 0.006737946999085467f;       // exp(-5)
    const float SPAN_  = (1.0f - START_) * (1.0f / 31.0f);
    const float Bq = 0.0625f * (1.0f - START_);
    const float BETA_ = 1.0f / (Bq * Bq);
    float ex = __expf(-ew);                            // ALPHA = 1
    #pragma unroll 4
    for (int r = 0; r < RNUM; ++r) {
        float t = ex - (START_ + SPAN_ * r);
        rbf[e * RNUM + r] = rc * __expf(-BETA_ * t * t);
    }
}

// Wn * mask(r_cut) * nbr_emb[z[src]]  scattered into msg[dst]
__global__ void msg_scatter_kernel(const float* __restrict__ Wn,
                                   const float* __restrict__ rcut,
                                   const int* __restrict__ eidx,
                                   const int* __restrict__ z,
                                   const float* __restrict__ nbr_emb,
                                   float* __restrict__ msg, int E) {
    int i = blockIdx.x * blockDim.x + threadIdx.x;
    if (i >= E * Hh) return;
    int e = i / Hh, hcol = i - e * Hh;
    int s = eidx[e], d = eidx[E + e];
    if (s == d) return;
    float w = Wn[i] * rcut[e] * nbr_emb[(size_t)z[s] * Hh + hcol];
    atomicAdd(&msg[(size_t)d * Hh + hcol], w);
}

__global__ void build_cat_kernel(const float* __restrict__ emb,
                                 const int* __restrict__ z,
                                 const float* __restrict__ msg,
                                 float* __restrict__ cat, int Nn) {
    int i = blockIdx.x * blockDim.x + threadIdx.x;
    if (i >= Nn * Hh) return;
    int n = i / Hh, hcol = i - n * Hh;
    cat[(size_t)n * (2 * Hh) + hcol]      = emb[(size_t)z[n] * Hh + hcol];
    cat[(size_t)n * (2 * Hh) + Hh + hcol] = msg[i];
}

__global__ void edge_attr_kernel(const float* __restrict__ x,
                                 const float* __restrict__ ep,
                                 const int* __restrict__ eidx,
                                 float* __restrict__ ea, int E) {
    int i = blockIdx.x * blockDim.x + threadIdx.x;
    if (i >= E * Hh) return;
    int e = i / Hh, hcol = i - e * Hh;
    int s = eidx[e], d = eidx[E + e];
    ea[i] = (x[(size_t)d * Hh + hcol] + x[(size_t)s * Hh + hcol]) * ep[i];
}

// LayerNorm over H=256: one block per row
__global__ void __launch_bounds__(Hh)
ln_kernel(const float* __restrict__ x, const float* __restrict__ g,
          const float* __restrict__ b, float* __restrict__ y) {
    __shared__ float red[Hh];
    int n = blockIdx.x, hcol = threadIdx.x;
    float v = x[(size_t)n * Hh + hcol];
    red[hcol] = v;
    __syncthreads();
    for (int s = Hh / 2; s > 0; s >>= 1) {
        if (hcol < s) red[hcol] += red[hcol + s];
        __syncthreads();
    }
    float mu = red[0] * (1.0f / Hh);
    __syncthreads();
    float xc = v - mu;
    red[hcol] = xc * xc;
    __syncthreads();
    for (int s = Hh / 2; s > 0; s >>= 1) {
        if (hcol < s) red[hcol] += red[hcol + s];
        __syncthreads();
    }
    float var = red[0] * (1.0f / Hh);
    y[(size_t)n * Hh + hcol] = xc * rsqrtf(var + 1e-5f) * g[hcol] + b[hcol];
}

// _vec_ln: per node, max/min over H of channel-group norms
__global__ void __launch_bounds__(Hh)
vecln_kernel(const float* __restrict__ vec, float* __restrict__ vl) {
    __shared__ float smx1[Hh], smn1[Hh], smx2[Hh], smn2[Hh];
    int n = blockIdx.x, hcol = threadIdx.x;
    const size_t base = (size_t)n * 8 * Hh + hcol;
    float v[8];
    #pragma unroll
    for (int c = 0; c < 8; ++c) v[c] = vec[base + c * Hh];
    float d1 = sqrtf(v[0] * v[0] + v[1] * v[1] + v[2] * v[2] + 1e-24f);
    d1 = fmaxf(d1, 1e-12f);
    float d2 = sqrtf(v[3] * v[3] + v[4] * v[4] + v[5] * v[5] +
                     v[6] * v[6] + v[7] * v[7] + 1e-24f);
    d2 = fmaxf(d2, 1e-12f);
    smx1[hcol] = d1; smn1[hcol] = d1; smx2[hcol] = d2; smn2[hcol] = d2;
    __syncthreads();
    for (int s = Hh / 2; s > 0; s >>= 1) {
        if (hcol < s) {
            smx1[hcol] = fmaxf(smx1[hcol], smx1[hcol + s]);
            smn1[hcol] = fminf(smn1[hcol], smn1[hcol + s]);
            smx2[hcol] = fmaxf(smx2[hcol], smx2[hcol + s]);
            smn2[hcol] = fminf(smn2[hcol], smn2[hcol + s]);
        }
        __syncthreads();
    }
    float mx1 = smx1[0], mn1 = smn1[0], mx2 = smx2[0], mn2 = smn2[0];
    float del1 = (mx1 - mn1 == 0.0f) ? 1.0f : (mx1 - mn1);
    float del2 = (mx2 - mn2 == 0.0f) ? 1.0f : (mx2 - mn2);
    float t1 = fmaxf((d1 - mn1) / del1, 0.0f) / d1;
    float t2 = fmaxf((d2 - mn2) / del2, 0.0f) / d2;
    #pragma unroll
    for (int c = 0; c < 3; ++c) vl[base + c * Hh] = t1 * v[c];
    #pragma unroll
    for (int c = 3; c < 8; ++c) vl[base + c * Hh] = t2 * v[c];
}

// vec_dot[n,o] = sum_c vec1[n,c,o] * vec2[n,c,o]  (vecWout rows: (n*8+c, 768))
__global__ void vecdot_kernel(const float* __restrict__ wout,
                              float* __restrict__ vdot, int Nn) {
    int i = blockIdx.x * blockDim.x + threadIdx.x;
    if (i >= Nn * Hh) return;
    int n = i / Hh, o = i - n * Hh;
    float s = 0.0f;
    #pragma unroll
    for (int c = 0; c < 8; ++c) {
        size_t row = (size_t)(n * 8 + c) * (3 * Hh);
        s += wout[row + o] * wout[row + Hh + o];
    }
    vdot[i] = s;
}

// attn[e,head] = silu( sum_hd q[dst]*k[src]*dk[e] ) * r_cut[e]
__global__ void attn_kernel(const float* __restrict__ q,
                            const float* __restrict__ k,
                            const float* __restrict__ dk,
                            const float* __restrict__ rcut,
                            const int* __restrict__ eidx,
                            float* __restrict__ attn, int E) {
    int i = blockIdx.x * blockDim.x + threadIdx.x;
    if (i >= E * NHEAD) return;
    int e = i / NHEAD, head = i - e * NHEAD;
    int s = eidx[e], d = eidx[E + e];
    const float* qp  = q  + (size_t)d * Hh + head * HDIM;
    const float* kp  = k  + (size_t)s * Hh + head * HDIM;
    const float* dkp = dk + (size_t)e * Hh + head * HDIM;
    float acc = 0.0f;
    #pragma unroll 8
    for (int t = 0; t < HDIM; ++t) acc += qp[t] * kp[t] * dkp[t];
    attn[i] = silu_f(acc) * rcut[e];
}

// vmsg[e,h] = v[src,h]*dv[e,h]*attn[e,h/32]; also x_agg[dst] += vmsg
__global__ void vmsg_kernel(const float* __restrict__ v,
                            const float* __restrict__ dv,
                            const float* __restrict__ attn,
                            const int* __restrict__ eidx,
                            float* __restrict__ vmsg,
                            float* __restrict__ x_agg, int E) {
    int i = blockIdx.x * blockDim.x + threadIdx.x;
    if (i >= E * Hh) return;
    int e = i / Hh, hcol = i - e * Hh;
    int s = eidx[e], d = eidx[E + e];
    float vm = v[(size_t)s * Hh + hcol] * dv[i] * attn[e * NHEAD + (hcol >> 5)];
    vmsg[i] = vm;
    atomicAdd(&x_agg[(size_t)d * Hh + hcol], vm);
}

// vec_agg[dst,c,h] += vl[src,c,h]*s1[e,h] + s2[e,h]*sph[e,c]
__global__ void vecagg_kernel(const float* __restrict__ vl,
                              const float* __restrict__ s12,
                              const float* __restrict__ sph,
                              const int* __restrict__ eidx,
                              float* __restrict__ vec_agg, int E) {
    int i = blockIdx.x * blockDim.x + threadIdx.x;
    if (i >= E * 8 * Hh) return;
    int e = i / (8 * Hh);
    int rem = i - e * 8 * Hh;
    int c = rem / Hh, hcol = rem - c * Hh;
    int s = eidx[e], d = eidx[E + e];
    float val = vl[(size_t)s * 8 * Hh + rem] * s12[(size_t)e * 2 * Hh + hcol] +
                s12[(size_t)e * 2 * Hh + Hh + hcol] * sph[e * 8 + c];
    atomicAdd(&vec_agg[(size_t)d * 8 * Hh + rem], val);
}

// x += vec_dot*o2 + o3
__global__ void xupd_kernel(float* __restrict__ x,
                            const float* __restrict__ vdot,
                            const float* __restrict__ o, int Nn) {
    int i = blockIdx.x * blockDim.x + threadIdx.x;
    if (i >= Nn * Hh) return;
    int n = i / Hh, hcol = i - n * Hh;
    size_t orow = (size_t)n * 3 * Hh;
    x[i] += vdot[i] * o[orow + Hh + hcol] + o[orow + 2 * Hh + hcol];
}

// vec += vec3*o1 + vec_agg
__global__ void vecupd_kernel(float* __restrict__ vec,
                              const float* __restrict__ wout,
                              const float* __restrict__ o,
                              const float* __restrict__ vec_agg, int Nn) {
    int i = blockIdx.x * blockDim.x + threadIdx.x;
    if (i >= Nn * 8 * Hh) return;
    int n = i / (8 * Hh);
    int rem = i - n * 8 * Hh;
    int c = rem / Hh, hcol = rem - c * Hh;
    float v3 = wout[(size_t)(n * 8 + c) * (3 * Hh) + 2 * Hh + hcol];
    vec[i] += v3 * o[(size_t)n * 3 * Hh + hcol] + vec_agg[i];
}

// edge_attr += f1*sum_c(w_t[dst]*w_s[src]) + f2*sum_c(t_t[dst]*t_s[src])
__global__ void edgeupd_kernel(float* __restrict__ ea,
                               const float* __restrict__ f12,
                               const float* __restrict__ w_t,
                               const float* __restrict__ w_s,
                               const float* __restrict__ t_t,
                               const float* __restrict__ t_s,
                               const int* __restrict__ eidx, int E) {
    int i = blockIdx.x * blockDim.x + threadIdx.x;
    if (i >= E * Hh) return;
    int e = i / Hh, hcol = i - e * Hh;
    int s = eidx[e], d = eidx[E + e];
    size_t db = (size_t)d * 8 * Hh + hcol;
    size_t sb = (size_t)s * 8 * Hh + hcol;
    float sw = 0.0f, st = 0.0f;
    #pragma unroll
    for (int c = 0; c < 8; ++c) {
        sw += w_t[db + c * Hh] * w_s[sb + c * Hh];
        st += t_t[db + c * Hh] * t_s[sb + c * Hh];
    }
    ea[i] += f12[(size_t)e * 2 * Hh + hcol] * sw +
             f12[(size_t)e * 2 * Hh + Hh + hcol] * st;
}

// =====================================================================
// Host orchestration
// =====================================================================
extern "C" void kernel_launch(void* const* d_in, const int* in_sizes, int n_in,
                              void* d_out, int out_size, void* d_ws, size_t ws_size,
                              hipStream_t stream) {
    const int*   z    = (const int*)d_in[0];
    const float* pos  = (const float*)d_in[1];
    const int*   eidx = (const int*)d_in[3];
    const int N = in_sizes[0];
    const int E = in_sizes[3] / 2;

    const float* emb         = (const float*)d_in[4];
    const float* nbr_emb     = (const float*)d_in[5];
    const float* nbr_dist_W  = (const float*)d_in[6];
    const float* nbr_dist_b  = (const float*)d_in[7];
    const float* nbr_comb_W  = (const float*)d_in[8];
    const float* nbr_comb_b  = (const float*)d_in[9];
    const float* edge_proj_W = (const float*)d_in[10];
    const float* edge_proj_b = (const float*)d_in[11];
    const float* ln_g  = (const float*)d_in[12];
    const float* ln_b  = (const float*)d_in[13];
    const float* vec_W = (const float*)d_in[14];
    const float* q_W   = (const float*)d_in[15];
    const float* q_b   = (const float*)d_in[16];
    const float* k_W   = (const float*)d_in[17];
    const float* k_b   = (const float*)d_in[18];
    const float* v_W   = (const float*)d_in[19];
    const float* v_b   = (const float*)d_in[20];
    const float* dk_W  = (const float*)d_in[21];
    const float* dk_b  = (const float*)d_in[22];
    const float* dv_W  = (const float*)d_in[23];
    const float* dv_b  = (const float*)d_in[24];
    const float* s_W   = (const float*)d_in[25];
    const float* s_b   = (const float*)d_in[26];
    const float* o_W   = (const float*)d_in[27];
    const float* o_b   = (const float*)d_in[28];
    const float* f_W   = (const float*)d_in[29];
    const float* f_b   = (const float*)d_in[30];
    const float* w_src_W = (const float*)d_in[31];
    const float* w_trg_W = (const float*)d_in[32];
    const float* t_src_W = (const float*)d_in[33];
    const float* t_trg_W = (const float*)d_in[34];
    const float* out_g = (const float*)d_in[35];
    const float* out_b = (const float*)d_in[36];

    // ---- workspace carving ----
    char* wsp = (char*)d_ws;
    size_t off = 0;
    auto alloc = [&](size_t elems) -> float* {
        float* p = (float*)(wsp + off);
        off += ((elems * sizeof(float) + 255) / 256) * 256;
        return p;
    };
    float* sph   = alloc((size_t)E * 8);
    float* rbf   = alloc((size_t)E * RNUM);
    float* rcut  = alloc((size_t)E);
    float* x     = alloc((size_t)N * Hh);
    float* vec   = alloc((size_t)N * 8 * Hh);
    float* ea    = alloc((size_t)E * Hh);
    float* msg   = alloc((size_t)N * Hh);
    float* cat   = alloc((size_t)N * 2 * Hh);
    float* Wn    = alloc((size_t)E * Hh);           // reused for edge_proj out
    float* xl    = alloc((size_t)N * Hh);
    float* vl    = alloc((size_t)N * 8 * Hh);
    float* qb    = alloc((size_t)N * Hh);
    float* kb    = alloc((size_t)N * Hh);
    float* vb    = alloc((size_t)N * Hh);
    float* un    = alloc((size_t)N * 32 * Hh);      // vecWout / w-t union
    float* vecWout = un;
    float* w_t = un;
    float* w_s = un + (size_t)N * 8 * Hh;
    float* t_t = un + (size_t)2 * N * 8 * Hh;
    float* t_s = un + (size_t)3 * N * 8 * Hh;
    float* vdot  = alloc((size_t)N * Hh);
    float* s12   = alloc((size_t)E * 2 * Hh);       // aliases dk|dv|f12
    float* dk    = s12;
    float* dv    = s12 + (size_t)E * Hh;
    float* f12   = s12;
    float* attn  = alloc((size_t)E * NHEAD);
    float* vmsg  = alloc((size_t)E * Hh);
    float* x_agg = alloc((size_t)N * Hh);
    float* vec_agg = alloc((size_t)N * 8 * Hh);
    float* obuf  = alloc((size_t)N * 3 * Hh);
    (void)ws_size; (void)n_in; (void)out_size;

    auto blocks = [](long long n) { return (unsigned)((n + 255) / 256); };

    // ---- preamble ----
    hipMemsetAsync(msg, 0, (size_t)N * Hh * 4, stream);
    hipMemsetAsync(vec, 0, (size_t)N * 8 * Hh * 4, stream);
    edge_geom_kernel<<<blocks(E), 256, 0, stream>>>(pos, eidx, E, sph, rbf, rcut);
    launch_gemm(rbf, nbr_dist_W, nbr_dist_b, Wn, E, Hh, RNUM, 0, stream);
    msg_scatter_kernel<<<blocks((long long)E * Hh), 256, 0, stream>>>(
        Wn, rcut, eidx, z, nbr_emb, msg, E);
    build_cat_kernel<<<blocks((long long)N * Hh), 256, 0, stream>>>(emb, z, msg, cat, N);
    launch_gemm(cat, nbr_comb_W, nbr_comb_b, x, N, Hh, 2 * Hh, 0, stream);
    launch_gemm(rbf, edge_proj_W, edge_proj_b, Wn, E, Hh, RNUM, 0, stream);
    edge_attr_kernel<<<blocks((long long)E * Hh), 256, 0, stream>>>(x, Wn, eidx, ea, E);

    // ---- layers ----
    for (int l = 0; l < LNUM; ++l) {
        ln_kernel<<<N, Hh, 0, stream>>>(x, ln_g + (size_t)l * Hh, ln_b + (size_t)l * Hh, xl);
        vecln_kernel<<<N, Hh, 0, stream>>>(vec, vl);
        launch_gemm(xl, q_W + (size_t)l * Hh * Hh, q_b + (size_t)l * Hh, qb, N, Hh, Hh, 0, stream);
        launch_gemm(xl, k_W + (size_t)l * Hh * Hh, k_b + (size_t)l * Hh, kb, N, Hh, Hh, 0, stream);
        launch_gemm(xl, v_W + (size_t)l * Hh * Hh, v_b + (size_t)l * Hh, vb, N, Hh, Hh, 0, stream);
        launch_gemm(ea, dk_W + (size_t)l * Hh * Hh, dk_b + (size_t)l * Hh, dk, E, Hh, Hh, 1, stream);
        launch_gemm(ea, dv_W + (size_t)l * Hh * Hh, dv_b + (size_t)l * Hh, dv, E, Hh, Hh, 1, stream);
        launch_gemm(vl, vec_W + (size_t)l * 3 * Hh * Hh, nullptr, vecWout, N * 8, 3 * Hh, Hh, 0, stream);
        vecdot_kernel<<<blocks((long long)N * Hh), 256, 0, stream>>>(vecWout, vdot, N);
        attn_kernel<<<blocks((long long)E * NHEAD), 256, 0, stream>>>(
            qb, kb, dk, rcut, eidx, attn, E);
        hipMemsetAsync(x_agg, 0, (size_t)N * Hh * 4, stream);
        vmsg_kernel<<<blocks((long long)E * Hh), 256, 0, stream>>>(
            vb, dv, attn, eidx, vmsg, x_agg, E);
        launch_gemm(vmsg, s_W + (size_t)l * 2 * Hh * Hh, s_b + (size_t)l * 2 * Hh,
                    s12, E, 2 * Hh, Hh, 1, stream);
        hipMemsetAsync(vec_agg, 0, (size_t)N * 8 * Hh * 4, stream);
        vecagg_kernel<<<blocks((long long)E * 8 * Hh), 256, 0, stream>>>(
            vl, s12, sph, eidx, vec_agg, E);
        launch_gemm(x_agg, o_W + (size_t)l * 3 * Hh * Hh, o_b + (size_t)l * 3 * Hh,
                    obuf, N, 3 * Hh, Hh, 0, stream);
        xupd_kernel<<<blocks((long long)N * Hh), 256, 0, stream>>>(x, vdot, obuf, N);
        vecupd_kernel<<<blocks((long long)N * 8 * Hh), 256, 0, stream>>>(
            vec, vecWout, obuf, vec_agg, N);
        if (l < LNUM - 1) {
            // w/t GEMMs may now overwrite vecWout (union region)
            launch_gemm(vl, w_trg_W + (size_t)l * Hh * Hh, nullptr, w_t, N * 8, Hh, Hh, 0, stream);
            launch_gemm(vl, w_src_W + (size_t)l * Hh * Hh, nullptr, w_s, N * 8, Hh, Hh, 0, stream);
            launch_gemm(vl, t_trg_W + (size_t)l * Hh * Hh, nullptr, t_t, N * 8, Hh, Hh, 0, stream);
            launch_gemm(vl, t_src_W + (size_t)l * Hh * Hh, nullptr, t_s, N * 8, Hh, Hh, 0, stream);
            launch_gemm(ea, f_W + (size_t)l * 2 * Hh * Hh, f_b + (size_t)l * 2 * Hh,
                        f12, E, 2 * Hh, Hh, 1, stream);
            edgeupd_kernel<<<blocks((long long)E * Hh), 256, 0, stream>>>(
                ea, f12, w_t, w_s, t_t, t_s, eidx, E);
        }
    }

    // ---- outputs: [ LN(x) | vec_ln(vec) ] ----
    float* out = (float*)d_out;
    ln_kernel<<<N, Hh, 0, stream>>>(x, out_g, out_b, out);
    vecln_kernel<<<N, Hh, 0, stream>>>(vec, out + (size_t)N * Hh);
}
